// Attention_16054587752918
// MI455X (gfx1250) — matrix operand order repaired and checked by reading, not compiled, not run on hardware
//
#include <hip/hip_runtime.h>
#include <hip/hip_bf16.h>
#include <math.h>

#define DIMV   768
#define NH     12
#define HD     64
#define SEQ    1024
#define BATCH  8
#define SCALEF 0.125f   // 64^-0.5

typedef __attribute__((ext_vector_type(16))) __bf16 bf16x16;
typedef __attribute__((ext_vector_type(8)))  __bf16 bf16x8;
typedef __attribute__((ext_vector_type(8)))  float  f32x8;
typedef int i32x4 __attribute__((vector_size(16)));   // matches builtin param type

#if defined(__HIP_DEVICE_COMPILE__) && \
    __has_builtin(__builtin_amdgcn_global_load_async_to_lds_b128) && \
    __has_builtin(__builtin_amdgcn_s_wait_asynccnt)
#define USE_ASYNC 1
#else
#define USE_ASYNC 0
#endif

#define GLB_AS __attribute__((address_space(1)))
#define LDS_AS __attribute__((address_space(3)))

static __device__ __forceinline__ f32x8 wmma_bf16(bf16x16 a, bf16x16 b, f32x8 c) {
  // D = A(16x32) * B(32x16) + C, f32 accum
  return __builtin_amdgcn_wmma_f32_16x16x32_bf16(false, a, false, b, (short)0, c,
                                                 false, false);
}

// A-matrix 16x32 bf16 fragment from row-major source (row stride `ld` elems).
// lanes 0-15: row M=lane, K=0..15; lanes 16-31: row M=lane-16, K=16..31
static __device__ __forceinline__ bf16x16 load_a_frag(const __bf16* base, int ld, int lane) {
  int r = lane & 15, hi = lane >> 4;
  const __bf16* p = base + r * ld + 16 * hi;
  union { bf16x16 v; bf16x8 h[2]; } u;
  u.h[0] = *(const bf16x8*)(p);       // K = 16*hi .. 16*hi+7
  u.h[1] = *(const bf16x8*)(p + 8);   // K = 16*hi+8 .. 16*hi+15
  return u.v;
}

// B-matrix 32x16 fragment where B[k][n] = src[n][k] (src row-major, stride ld):
// lanes 0-15: N=lane, K=0..15; lanes 16-31: N=lane-16, K=16..31 (contiguous 32B run)
static __device__ __forceinline__ bf16x16 load_bt_frag(const __bf16* base, int ld, int lane) {
  int n = lane & 15, hi = lane >> 4;
  return *(const bf16x16*)(base + n * ld + 16 * hi);
}

// 16-byte global -> LDS stage: async (ASYNCcnt path) when available.
static __device__ __forceinline__ void stage16(const __bf16* g, __bf16* l) {
#if USE_ASYNC
  __builtin_amdgcn_global_load_async_to_lds_b128(
      (GLB_AS i32x4*)(unsigned long long)(uintptr_t)g,
      (LDS_AS i32x4*)(unsigned)(uintptr_t)l, 0, 0);
#else
  *(bf16x8*)l = *(const bf16x8*)g;
#endif
}

template <int N>
static __device__ __forceinline__ void wait_async() {
#if USE_ASYNC
  __builtin_amdgcn_s_wait_asynccnt(N);
#endif
}

// ---------------------------------------------------------------- convert
__global__ void cvt_f32_bf16(const float* __restrict__ in, __bf16* __restrict__ out, int n) {
  int i = blockIdx.x * blockDim.x + threadIdx.x;
  int stride = gridDim.x * blockDim.x;
  for (; i < n; i += stride) out[i] = (__bf16)in[i];
}

// ---------------------------------------------------------------- QKV GEMM
// qkv[m][n] = sum_k xb[m][k]*wb[n][k] + bias[n]; block tile 128x128, wave 32x64,
// double-buffered LDS staged with async global->LDS.
__global__ void __launch_bounds__(256) qkv_gemm(const __bf16* __restrict__ xb,
                                                const __bf16* __restrict__ wb,
                                                const float* __restrict__ bias,
                                                __bf16* __restrict__ Qb,
                                                __bf16* __restrict__ Kb,
                                                __bf16* __restrict__ Vtb) {
  __shared__ __align__(64) __bf16 As[2][128 * 32];
  __shared__ __align__(64) __bf16 Bs[2][128 * 32];
  const int tid = threadIdx.x;
  const int lane = tid & 31, wid = tid >> 5;
  const int hi = lane >> 4, nl = lane & 15;
  const int m0 = blockIdx.y * 128, n0 = blockIdx.x * 128;
  const int wm = (wid & 3) * 32, wn = (wid >> 2) * 64;
  // staging: each thread moves 4x16B per K-step (rows crow and crow+64 of A and B)
  const int crow = tid >> 2, cq = (tid & 3) * 8;

  f32x8 acc[2][4] = {};
  const int NST = DIMV / 32;

  auto stage = [&](int buf, int k0) {
    stage16(xb + (size_t)(m0 + crow) * DIMV + k0 + cq,      &As[buf][crow * 32 + cq]);
    stage16(xb + (size_t)(m0 + 64 + crow) * DIMV + k0 + cq, &As[buf][(64 + crow) * 32 + cq]);
    stage16(wb + (size_t)(n0 + crow) * DIMV + k0 + cq,      &Bs[buf][crow * 32 + cq]);
    stage16(wb + (size_t)(n0 + 64 + crow) * DIMV + k0 + cq, &Bs[buf][(64 + crow) * 32 + cq]);
  };
  auto compute = [&](int buf) {
    bf16x16 a0 = load_a_frag(&As[buf][wm * 32], 32, lane);
    bf16x16 a1 = load_a_frag(&As[buf][(wm + 16) * 32], 32, lane);
#pragma unroll
    for (int j = 0; j < 4; ++j) {
      bf16x16 bj = load_bt_frag(&Bs[buf][(wn + j * 16) * 32], 32, lane);
      acc[0][j] = wmma_bf16(a0, bj, acc[0][j]);
      acc[1][j] = wmma_bf16(a1, bj, acc[1][j]);
    }
  };

  stage(0, 0);
  for (int kt = 0; kt < NST - 1; ++kt) {
    stage((kt + 1) & 1, (kt + 1) * 32);
    wait_async<4>();  // in-order completion: stage kt done, stage kt+1 in flight
    __syncthreads();
    compute(kt & 1);
    __syncthreads();
  }
  wait_async<0>();
  __syncthreads();
  compute((NST - 1) & 1);

#pragma unroll
  for (int i = 0; i < 2; ++i)
#pragma unroll
    for (int j = 0; j < 4; ++j)
#pragma unroll
      for (int r = 0; r < 8; ++r) {
        int m = m0 + wm + i * 16 + r + 8 * hi;
        int n = n0 + wn + j * 16 + nl;
        float v = acc[i][j][r] + bias[n];
        int t = n / DIMV, rem = n - t * DIMV;
        int h = rem >> 6, d = rem & 63;
        int b = m >> 10, s = m & 1023;
        int bh = b * NH + h;
        if (t == 0)
          Qb[((size_t)bh * SEQ + s) * HD + d] = (__bf16)(v * SCALEF);
        else if (t == 1)
          Kb[((size_t)bh * SEQ + s) * HD + d] = (__bf16)v;
        else
          Vtb[((size_t)bh * HD + d) * SEQ + s] = (__bf16)v;
      }
}

// ---------------------------------------------------------------- attention
// One wave = 16 queries, flash-style online softmax over 32-key chunks.
__global__ void __launch_bounds__(256) attn_kernel(const __bf16* __restrict__ Qb,
                                                   const __bf16* __restrict__ Kb,
                                                   const __bf16* __restrict__ Vtb,
                                                   __bf16* __restrict__ Ob) {
  __shared__ __align__(64) __bf16 Pst[8][16 * 32];  // per-wave P bounce buffer
  const int tid = threadIdx.x, lane = tid & 31, wid = tid >> 5;
  const int hi = lane >> 4, nl = lane & 15;
  const int bh = blockIdx.y;
  const int b = bh / NH, h = bh - b * NH;
  const int q0 = blockIdx.x * 128 + wid * 16;
  const __bf16* Qp = Qb + ((size_t)bh * SEQ + q0) * HD;
  const __bf16* Kp = Kb + (size_t)bh * SEQ * HD;
  const __bf16* Vp = Vtb + (size_t)bh * HD * SEQ;

  bf16x16 qa0 = load_a_frag(Qp, HD, lane);       // d = 0..31
  bf16x16 qa1 = load_a_frag(Qp + 32, HD, lane);  // d = 32..63

  f32x8 O[4] = {};
  float mrow[8], lrow[8];
#pragma unroll
  for (int r = 0; r < 8; ++r) { mrow[r] = -1e30f; lrow[r] = 0.0f; }
  __bf16* Pw = &Pst[wid][0];

  for (int j0 = 0; j0 < SEQ; j0 += 32) {
    const __bf16* Kt = Kp + (size_t)j0 * HD;
    // scores (Q already carries the 1/sqrt(d) scale)
    f32x8 s0 = {}, s1 = {};
    s0 = wmma_bf16(qa0, load_bt_frag(Kt, HD, lane), s0);
    s0 = wmma_bf16(qa1, load_bt_frag(Kt + 32, HD, lane), s0);
    s1 = wmma_bf16(qa0, load_bt_frag(Kt + 16 * HD, HD, lane), s1);
    s1 = wmma_bf16(qa1, load_bt_frag(Kt + 16 * HD + 32, HD, lane), s1);

#pragma unroll
    for (int r = 0; r < 8; ++r) {
      float v0 = s0[r], v1 = s1[r];
      float mx = fmaxf(v0, v1);
      mx = fmaxf(mx, __shfl_xor(mx, 1, 32));
      mx = fmaxf(mx, __shfl_xor(mx, 2, 32));
      mx = fmaxf(mx, __shfl_xor(mx, 4, 32));
      mx = fmaxf(mx, __shfl_xor(mx, 8, 32));
      float mnew = fmaxf(mrow[r], mx);
      float corr = __expf(mrow[r] - mnew);
      float p0 = __expf(v0 - mnew);
      float p1 = __expf(v1 - mnew);
      float ps = p0 + p1;
      ps += __shfl_xor(ps, 1, 32);
      ps += __shfl_xor(ps, 2, 32);
      ps += __shfl_xor(ps, 4, 32);
      ps += __shfl_xor(ps, 8, 32);
      lrow[r] = lrow[r] * corr + ps;
      mrow[r] = mnew;
#pragma unroll
      for (int ct = 0; ct < 4; ++ct) O[ct][r] *= corr;
      // D-layout -> LDS (row-major 16x32) for re-load in A-layout
      Pw[(r + 8 * hi) * 32 + nl]      = (__bf16)p0;
      Pw[(r + 8 * hi) * 32 + 16 + nl] = (__bf16)p1;
    }
    // Pst[wid] is wave-private; LDS ops within a wave complete in order, so a
    // compiler-only fence suffices (no cross-wave barrier needed).
    asm volatile("" ::: "memory");
    bf16x16 pa = load_a_frag(Pw, 32, lane);
#pragma unroll
    for (int ct = 0; ct < 4; ++ct) {
      // B[k][n] = V[j0+k][ct*16+n] = Vt[(ct*16+n)*SEQ + j0+k] -> contiguous
      bf16x16 vb = load_bt_frag(Vp + (size_t)(ct * 16) * SEQ + j0, SEQ, lane);
      O[ct] = wmma_bf16(pa, vb, O[ct]);
    }
    asm volatile("" ::: "memory");
  }
#pragma unroll
  for (int r = 0; r < 8; ++r) {
    float inv = 1.0f / lrow[r];
    int q = q0 + r + 8 * hi;
#pragma unroll
    for (int ct = 0; ct < 4; ++ct) {
      float v = O[ct][r] * inv;
      Ob[((size_t)(b * SEQ + q)) * DIMV + h * HD + ct * 16 + nl] = (__bf16)v;
    }
  }
}

// ---------------------------------------------------------------- proj GEMM
__global__ void __launch_bounds__(256) proj_gemm(const __bf16* __restrict__ ab,
                                                 const __bf16* __restrict__ pw,
                                                 const float* __restrict__ bias,
                                                 float* __restrict__ out) {
  __shared__ __align__(64) __bf16 As[2][128 * 32];
  __shared__ __align__(64) __bf16 Bs[2][128 * 32];
  const int tid = threadIdx.x;
  const int lane = tid & 31, wid = tid >> 5;
  const int hi = lane >> 4, nl = lane & 15;
  const int m0 = blockIdx.y * 128, n0 = blockIdx.x * 128;
  const int wm = (wid & 3) * 32, wn = (wid >> 2) * 64;
  const int crow = tid >> 2, cq = (tid & 3) * 8;

  f32x8 acc[2][4] = {};
  const int NST = DIMV / 32;

  auto stage = [&](int buf, int k0) {
    stage16(ab + (size_t)(m0 + crow) * DIMV + k0 + cq,      &As[buf][crow * 32 + cq]);
    stage16(ab + (size_t)(m0 + 64 + crow) * DIMV + k0 + cq, &As[buf][(64 + crow) * 32 + cq]);
    stage16(pw + (size_t)(n0 + crow) * DIMV + k0 + cq,      &Bs[buf][crow * 32 + cq]);
    stage16(pw + (size_t)(n0 + 64 + crow) * DIMV + k0 + cq, &Bs[buf][(64 + crow) * 32 + cq]);
  };
  auto compute = [&](int buf) {
    bf16x16 a0 = load_a_frag(&As[buf][wm * 32], 32, lane);
    bf16x16 a1 = load_a_frag(&As[buf][(wm + 16) * 32], 32, lane);
#pragma unroll
    for (int j = 0; j < 4; ++j) {
      bf16x16 bj = load_bt_frag(&Bs[buf][(wn + j * 16) * 32], 32, lane);
      acc[0][j] = wmma_bf16(a0, bj, acc[0][j]);
      acc[1][j] = wmma_bf16(a1, bj, acc[1][j]);
    }
  };

  stage(0, 0);
  for (int kt = 0; kt < NST - 1; ++kt) {
    stage((kt + 1) & 1, (kt + 1) * 32);
    wait_async<4>();
    __syncthreads();
    compute(kt & 1);
    __syncthreads();
  }
  wait_async<0>();
  __syncthreads();
  compute((NST - 1) & 1);

#pragma unroll
  for (int i = 0; i < 2; ++i)
#pragma unroll
    for (int j = 0; j < 4; ++j)
#pragma unroll
      for (int r = 0; r < 8; ++r) {
        int m = m0 + wm + i * 16 + r + 8 * hi;
        int n = n0 + wn + j * 16 + nl;
        out[(size_t)m * DIMV + n] = acc[i][j][r] + bias[n];
      }
}

// ---------------------------------------------------------------- launch
extern "C" void kernel_launch(void* const* d_in, const int* in_sizes, int n_in,
                              void* d_out, int out_size, void* d_ws, size_t ws_size,
                              hipStream_t stream) {
  (void)in_sizes; (void)n_in; (void)out_size; (void)ws_size;
  const float* x      = (const float*)d_in[0];
  const float* qkv_w  = (const float*)d_in[1];
  const float* qkv_b  = (const float*)d_in[2];
  const float* proj_w = (const float*)d_in[3];
  const float* proj_b = (const float*)d_in[4];
  float* out = (float*)d_out;

  char* ws = (char*)d_ws;
  size_t off = 0;
  auto alloc = [&](size_t elems) -> __bf16* {
    __bf16* p = (__bf16*)(ws + off);
    off += (elems * sizeof(__bf16) + 255) & ~(size_t)255;
    return p;
  };
  const size_t NTOK = (size_t)BATCH * SEQ;           // 8192
  __bf16* xb  = alloc(NTOK * DIMV);                  // x bf16
  __bf16* wb  = alloc((size_t)3 * DIMV * DIMV);      // qkv_w bf16
  __bf16* pb  = alloc((size_t)DIMV * DIMV);          // proj_w bf16
  __bf16* Qb  = alloc(NTOK * DIMV);                  // [b,h,s,d] (pre-scaled)
  __bf16* Kb  = alloc(NTOK * DIMV);                  // [b,h,s,d]
  __bf16* Vtb = alloc(NTOK * DIMV);                  // [b,h,d,s]
  __bf16* ab  = alloc(NTOK * DIMV);                  // attn out [b,s,h*d]

  cvt_f32_bf16<<<1024, 256, 0, stream>>>(x, xb, (int)(NTOK * DIMV));
  cvt_f32_bf16<<<1024, 256, 0, stream>>>(qkv_w, wb, 3 * DIMV * DIMV);
  cvt_f32_bf16<<<1024, 256, 0, stream>>>(proj_w, pb, DIMV * DIMV);

  qkv_gemm<<<dim3(3 * DIMV / 128, NTOK / 128), 256, 0, stream>>>(xb, wb, qkv_b,
                                                                 Qb, Kb, Vtb);
  attn_kernel<<<dim3(SEQ / 128, BATCH * NH), 256, 0, stream>>>(Qb, Kb, Vtb, ab);
  proj_gemm<<<dim3(DIMV / 128, NTOK / 128), 256, 0, stream>>>(ab, pb, proj_b, out);
}
